// Bidirectional_GRU_Decoder_21234318311667
// MI455X (gfx1250) — compile-verified, matrix operand order checked by reading, not run on hardware
//
#include <hip/hip_runtime.h>
#include <stdint.h>
#include <stddef.h>

// Problem constants (match reference)
#define LL 2
#define TT 1024
#define BB 64
#define HH 512
#define KK 512     // DIN == H == 512, so one GEMM shape everywhere
#define GG 1536    // 3*H gate dimension

typedef __attribute__((ext_vector_type(16))) __bf16 v16bf;
typedef __attribute__((ext_vector_type(8)))  float  v8f;

union FragB { v16bf v; uint4 q[2]; };

__device__ __forceinline__ unsigned short f2bf(float f) {
  unsigned u = __float_as_uint(f);
  unsigned r = u + 0x7FFFu + ((u >> 16) & 1u);   // round-to-nearest-even
  return (unsigned short)(r >> 16);
}
__device__ __forceinline__ float bf2f(unsigned short h) {
  return __uint_as_float(((unsigned)h) << 16);
}
__device__ __forceinline__ float sigm(float x)      { return 1.0f / (1.0f + __expf(-x)); }
__device__ __forceinline__ float tanh_fast(float x) { return 2.0f / (1.0f + __expf(-2.0f * x)) - 1.0f; }

// CDNA5 async global->LDS copy (ASYNCcnt path, no VGPR round-trip).
// VDST = wave-relative LDS byte address, VADDR = 64-bit global address.
__device__ __forceinline__ void async_ld_b128(uint32_t lds_off, const void* gaddr) {
  asm volatile("global_load_async_to_lds_b128 %0, %1, off"
               :: "v"(lds_off), "v"((unsigned long long)(uintptr_t)gaddr)
               : "memory");
}
__device__ __forceinline__ void wait_async0() {
#if __has_builtin(__builtin_amdgcn_s_wait_asynccnt)
  __builtin_amdgcn_s_wait_asynccnt(0);
#else
  asm volatile("s_wait_asynccnt 0x0" ::: "memory");
#endif
}

// ---------------------------------------------------------------------------
// f32 -> bf16 flat convert (weights)
__global__ void k_cvt(const float* __restrict__ src, unsigned short* __restrict__ dst, int n) {
  int i = blockIdx.x * blockDim.x + threadIdx.x;
  if (i < n) dst[i] = f2bf(src[i]);
}

// x [B,T,512] f32 -> xb [T,B,512] bf16 (reference transposes to time-major)
__global__ void k_cvt_x(const float* __restrict__ x, unsigned short* __restrict__ xb) {
  size_t i = (size_t)blockIdx.x * blockDim.x + threadIdx.x;   // over B*T*512
  int    d  = (int)(i & 511);
  size_t bt = i >> 9;
  int    t  = (int)(bt & (TT - 1));
  int    b  = (int)(bt >> 10);
  xb[(((size_t)t * BB + b) << 9) + d] = f2bf(x[i]);
}

// h0 init from encoder_h[layer,:, dir*H : dir*H+H] into per-dir h buffers
__global__ void k_init_h(const float* __restrict__ enc, float* __restrict__ hF,
                         unsigned short* __restrict__ hB, int layer) {
  int i = blockIdx.x * blockDim.x + threadIdx.x;      // over 2*64*512
  if (i >= 2 * BB * HH) return;
  int c = i & 511;
  int b = (i >> 9) & 63;
  int dir = i >> 15;
  float v = enc[(size_t)layer * BB * 2 * HH + (size_t)b * 2 * HH + dir * HH + c];
  hF[i] = v;
  hB[i] = f2bf(v);
}

// ---------------------------------------------------------------------------
// gx = A * W^T + bias   (A: [T*64,512] bf16 rows; W: [1536,512] bf16; gx bf16)
// Block tile: M=64 (one timestep's 64 batch rows) x N=256. 8 waves, each 32x64.
__global__ void __launch_bounds__(256) k_gemm_gx(
    const unsigned short* __restrict__ A,
    const unsigned short* __restrict__ W,
    const float* __restrict__ bias,
    unsigned short* __restrict__ gx,
    int rev)
{
  const int tt    = blockIdx.y;
  const int t_src = rev ? (TT - 1 - tt) : tt;
  const int lane  = threadIdx.x & 31;
  const int wave  = threadIdx.x >> 5;
  const int wm    = wave & 1;        // M offset 0/32
  const int wn    = wave >> 1;       // N offset 0/64/128/192
  const int lr    = lane & 15;
  const int hi    = lane >> 4;       // K-half select per documented 16-bit A layout
  const int nbase = blockIdx.x * 256 + wn * 64;

  v8f acc[2][4];
#pragma unroll
  for (int i = 0; i < 2; i++)
#pragma unroll
    for (int j = 0; j < 4; j++)
      acc[i][j] = (v8f){0.f,0.f,0.f,0.f,0.f,0.f,0.f,0.f};

  const unsigned short* Arow0 = A + (size_t)t_src * BB * KK;

#pragma unroll 1
  for (int k = 0; k < KK; k += 32) {
    FragB fa[2];
#pragma unroll
    for (int mt = 0; mt < 2; mt++) {
      const unsigned short* p = Arow0 + (size_t)(wm * 32 + mt * 16 + lr) * KK + k + hi * 8;
      fa[mt].q[0] = *(const uint4*)(p);
      fa[mt].q[1] = *(const uint4*)(p + 16);
    }
    FragB fb[4];
#pragma unroll
    for (int nt = 0; nt < 4; nt++) {
      const unsigned short* p = W + (size_t)(nbase + nt * 16 + lr) * KK + k + hi * 8;
      fb[nt].q[0] = *(const uint4*)(p);
      fb[nt].q[1] = *(const uint4*)(p + 16);
    }
#pragma unroll
    for (int mt = 0; mt < 2; mt++)
#pragma unroll
      for (int nt = 0; nt < 4; nt++)
        acc[mt][nt] = __builtin_amdgcn_wmma_f32_16x16x32_bf16(
            false, fa[mt].v, false, fb[nt].v, (short)0, acc[mt][nt], false, false);
  }

  // Epilogue: +bias, convert to bf16, store. D layout: lane holds col n, rows hi*8+v.
#pragma unroll
  for (int mt = 0; mt < 2; mt++) {
#pragma unroll
    for (int nt = 0; nt < 4; nt++) {
      int n = nbase + nt * 16 + lr;
      float bv = bias[n];
      size_t rowbase = (size_t)tt * BB + wm * 32 + mt * 16 + hi * 8;
#pragma unroll
      for (int v = 0; v < 8; v++)
        gx[(rowbase + v) * GG + n] = f2bf(acc[mt][nt][v] + bv);
    }
  }
}

// ---------------------------------------------------------------------------
// Persistent GRU scan. Grid = 16 WGs; WG w owns hidden cols [w*32, w*32+32),
// computing all three gates for those columns each step.
// Dynamic LDS (160 KB of the WGP's 320 KB):
//   [0,      64 KB)  h state, re-staged each step via global_load_async_to_lds_b128
//   [64 KB, 160 KB)  this WG's Whh slice (96 rows x 512 K bf16), staged once
// f32 hidden state lives in registers (each lane owns its 8 (b,c) elements).
__global__ void __launch_bounds__(256) k_scan(
    const unsigned short* __restrict__ gx,    // [T*64,1536] bf16 (bias folded)
    const unsigned short* __restrict__ Whh,   // [1536,512] bf16
    const float* __restrict__ bhh,            // [1536] f32
    const float* __restrict__ hF,             // [64,512] f32 (init only)
    unsigned short* __restrict__ hB,          // [64,512] bf16 (cross-WG state)
    unsigned short* __restrict__ yOut,        // [T*64,512] bf16 (next-layer input)
    float* __restrict__ outMain,              // [B,T,1024] f32 (last layer only)
    float* __restrict__ outH,                 // [L,B,1024] f32
    unsigned int* __restrict__ sync,
    int dir, int layer, int lastLayer)
{
  extern __shared__ __attribute__((aligned(16))) char smem[];
  unsigned short* hS = (unsigned short*)smem;             // 65536 B
  unsigned short* wS = (unsigned short*)(smem + 65536);   // 98304 B

  const int lane = threadIdx.x & 31;
  const int wave = threadIdx.x >> 5;
  const int mt   = wave & 3;                  // M tile 0..3 (batch 16-row groups)
  const int nt   = wave >> 2;                 // N tile 0..1 within 32-col slice
  const int lr   = lane & 15;
  const int hi   = lane >> 4;
  const int c0   = blockIdx.x * 32;
  const int c    = c0 + nt * 16 + lr;         // hidden column this lane owns

  const float bR = bhh[c], bZ = bhh[512 + c], bN = bhh[1024 + c];

  // Stage this WG's Whh slice into LDS once: rows rl = g*32 + (c-c0), 6144 uint4.
#pragma unroll
  for (int i = 0; i < 24; ++i) {
    int e  = threadIdx.x + i * 256;           // uint4 index
    int rl = e >> 6;                          // 0..95
    int g  = rl >> 5, cl = rl & 31;
    const unsigned short* src = Whh + (size_t)(g * HH + c0 + cl) * KK + (e & 63) * 8;
    async_ld_b128((uint32_t)(uintptr_t)(wS + (size_t)e * 8), src);
  }

  // f32 hidden state in registers: lane owns (b = mt*16+hi*8+v, c)
  float hreg[8];
#pragma unroll
  for (int v = 0; v < 8; v++)
    hreg[v] = hF[(size_t)(mt * 16 + hi * 8 + v) * HH + c];

  const unsigned short* wRl = wS + (size_t)( 0 + nt * 16 + lr) * KK;
  const unsigned short* wZl = wS + (size_t)(32 + nt * 16 + lr) * KK;
  const unsigned short* wNl = wS + (size_t)(64 + nt * 16 + lr) * KK;

  for (int tt = 0; tt < TT; ++tt) {
    // Stage h (bf16, 64 KB = 4096 uint4) into LDS via async path
#pragma unroll
    for (int i = 0; i < 16; ++i) {
      int e = threadIdx.x + i * 256;
      async_ld_b128((uint32_t)(uintptr_t)(hS + (size_t)e * 8), hB + (size_t)e * 8);
    }
    wait_async0();
    __syncthreads();

    v8f aR = (v8f){0.f,0.f,0.f,0.f,0.f,0.f,0.f,0.f};
    v8f aZ = aR, aN = aR;

#pragma unroll 1
    for (int k = 0; k < KK; k += 32) {
      FragB fa;
      const unsigned short* p = &hS[(mt * 16 + lr) * KK + k + hi * 8];
      fa.q[0] = *(const uint4*)p;
      fa.q[1] = *(const uint4*)(p + 16);
      FragB fr, fz, fn;
      fr.q[0] = *(const uint4*)(wRl + k + hi * 8); fr.q[1] = *(const uint4*)(wRl + k + hi * 8 + 16);
      fz.q[0] = *(const uint4*)(wZl + k + hi * 8); fz.q[1] = *(const uint4*)(wZl + k + hi * 8 + 16);
      fn.q[0] = *(const uint4*)(wNl + k + hi * 8); fn.q[1] = *(const uint4*)(wNl + k + hi * 8 + 16);
      aR = __builtin_amdgcn_wmma_f32_16x16x32_bf16(false, fa.v, false, fr.v, (short)0, aR, false, false);
      aZ = __builtin_amdgcn_wmma_f32_16x16x32_bf16(false, fa.v, false, fz.v, (short)0, aZ, false, false);
      aN = __builtin_amdgcn_wmma_f32_16x16x32_bf16(false, fa.v, false, fn.v, (short)0, aN, false, false);
    }

    // Gate math + state update for the 8 (b,c) elements this lane owns
    size_t gxrow = (size_t)tt * BB;
#pragma unroll
    for (int v = 0; v < 8; v++) {
      int b = mt * 16 + hi * 8 + v;
      size_t g = (gxrow + b) * GG;
      float xr = bf2f(gx[g + c]);
      float xz = bf2f(gx[g + 512 + c]);
      float xn = bf2f(gx[g + 1024 + c]);
      float r  = sigm(xr + aR[v] + bR);
      float z  = sigm(xz + aZ[v] + bZ);
      float n  = tanh_fast(xn + r * (aN[v] + bN));
      float hn = (1.0f - z) * n + z * hreg[v];
      hreg[v] = hn;
      hB[(size_t)b * HH + c]     = f2bf(hn);
      yOut[(gxrow + b) * HH + c] = f2bf(hn);
      if (lastLayer) {
        int treal = dir ? (TT - 1 - tt) : tt;
        outMain[((size_t)b * TT + treal) * 1024 + dir * HH + c] = hn;
      }
      if (tt == TT - 1)
        outH[((size_t)layer * BB + b) * 1024 + dir * HH + c] = hn;
    }

    if (tt + 1 < TT)   // warm GL2 for next step's gx lines (global_prefetch_b8)
      __builtin_prefetch(&gx[((size_t)(tt + 1) * BB + mt * 16 + hi * 8) * GG + c], 0, 1);

    // Device-wide step barrier (16 WGs)
    __syncthreads();
    __threadfence();
    if (threadIdx.x == 0) {
      atomicAdd(sync, 1u);
      unsigned target = 16u * (unsigned)(tt + 1);
      while (__hip_atomic_load(sync, __ATOMIC_RELAXED, __HIP_MEMORY_SCOPE_AGENT) < target)
        __builtin_amdgcn_s_sleep(1);
    }
    __syncthreads();
    __threadfence();
  }
}

// ---------------------------------------------------------------------------
extern "C" void kernel_launch(void* const* d_in, const int* in_sizes, int n_in,
                              void* d_out, int out_size, void* d_ws, size_t ws_size,
                              hipStream_t stream) {
  (void)in_sizes; (void)n_in; (void)out_size;

  const float* x     = (const float*)d_in[0];
  const float* enc   = (const float*)d_in[1];
  const float* Wih_f = (const float*)d_in[2];
  const float* Whh_f = (const float*)d_in[3];
  const float* bih_f = (const float*)d_in[4];
  const float* bhh_f = (const float*)d_in[5];
  const float* Wih_b = (const float*)d_in[6];
  const float* Whh_b = (const float*)d_in[7];
  const float* bih_b = (const float*)d_in[8];
  const float* bhh_b = (const float*)d_in[9];

  // Workspace carve-up (~416 MB)
  char* ws = (char*)d_ws;
  size_t off = 0;
  auto carve = [&](size_t bytes) -> char* {
    char* p = ws + off;
    off += (bytes + 255) & ~(size_t)255;
    return p;
  };
  unsigned short* xb   = (unsigned short*)carve((size_t)TT * BB * KK * 2);        // 67.1 MB
  unsigned short* wih  = (unsigned short*)carve((size_t)2 * LL * GG * KK * 2);    // 6.3 MB
  unsigned short* whh  = (unsigned short*)carve((size_t)2 * LL * GG * KK * 2);    // 6.3 MB
  unsigned short* gxb  = (unsigned short*)carve((size_t)TT * BB * GG * 2);        // 201 MB
  unsigned short* ybuf = (unsigned short*)carve((size_t)2 * TT * BB * HH * 2);    // 134 MB
  float*          hF   = (float*)carve((size_t)2 * BB * HH * 4);
  unsigned short* hBf  = (unsigned short*)carve((size_t)2 * BB * HH * 2);
  unsigned int*   sync = (unsigned int*)carve(256);
  if (off > ws_size) return;   // insufficient scratch; bail deterministically

  float* outMain = (float*)d_out;
  float* outH    = (float*)d_out + (size_t)BB * TT * 2 * HH;

  hipMemsetAsync(sync, 0, 256, stream);

  // Precision conversion
  k_cvt_x<<<(TT * BB * KK) / 256, 256, 0, stream>>>(x, xb);
  const int WELEMS = LL * GG * KK;            // 1,572,864 per direction
  k_cvt<<<WELEMS / 256, 256, 0, stream>>>(Wih_f, wih,                       WELEMS);
  k_cvt<<<WELEMS / 256, 256, 0, stream>>>(Wih_b, wih + (size_t)WELEMS,      WELEMS);
  k_cvt<<<WELEMS / 256, 256, 0, stream>>>(Whh_f, whh,                       WELEMS);
  k_cvt<<<WELEMS / 256, 256, 0, stream>>>(Whh_b, whh + (size_t)WELEMS,      WELEMS);

  const size_t scanLds = 64 * 1024 + 96 * 1024;   // 160 KB of the WGP's 320 KB

  for (int layer = 0; layer < LL; ++layer) {
    k_init_h<<<(2 * BB * HH) / 256, 256, 0, stream>>>(enc, hF, hBf, layer);
    for (int dir = 0; dir < 2; ++dir) {
      const unsigned short* Ain  = (layer == 0) ? xb : (ybuf + (size_t)dir * TT * BB * HH);
      int rev = (layer == 0 && dir == 1);
      const unsigned short* wihp = wih + ((size_t)dir * LL + layer) * GG * KK;
      const unsigned short* whhp = whh + ((size_t)dir * LL + layer) * GG * KK;
      const float* bihp = (dir ? bih_b : bih_f) + (size_t)layer * GG;
      const float* bhhp = (dir ? bhh_b : bhh_f) + (size_t)layer * GG;

      k_gemm_gx<<<dim3(GG / 256, TT), 256, 0, stream>>>(Ain, wihp, bihp, gxb, rev);

      k_scan<<<16, 256, scanLds, stream>>>(gxb, whhp, bhhp,
                                           hF + (size_t)dir * BB * HH,
                                           hBf + (size_t)dir * BB * HH,
                                           ybuf + (size_t)dir * TT * BB * HH,
                                           (layer == LL - 1) ? outMain : nullptr,
                                           outH,
                                           sync + (size_t)(layer * 2 + dir) * 16,
                                           dir, layer, (layer == LL - 1) ? 1 : 0);
    }
  }
}
